// MambaDecoderLayer_15393162789180
// MI455X (gfx1250) — matrix operand order repaired and checked by reading, not compiled, not run on hardware
//
#include <hip/hip_runtime.h>
#include <hip/hip_bf16.h>

typedef __attribute__((ext_vector_type(16))) _Float16 v16h;
typedef __attribute__((ext_vector_type(8)))  float    v8f;

#define D_MODEL 512
#define D_INNER 1024
#define D_STATE 16
#define DT_RANK 32
#define D_CONV  4
#define D_FFN   2048
#define BATCH   16
#define L_TGT   512
#define L_SRC   1024
#define L_CAT   (L_SRC + L_TGT)
#define EPS_LN  1e-6f

// ---------------------------------------------------------------------------
// Weight packing: W (K x N, f32, row-major) -> f16 WMMA B-fragments.
// Layout: Wp[((kc*(N/16) + tn)*32 + lane)*16 + j] =
//   W[kc*32 + (j>>3)*16 + (lane>>4)*8 + (j&7)][tn*16 + (lane&15)]
// so each lane's B fragment for (k-chunk kc, 16-col tile tn) is 16 contiguous
// halfs (32 bytes). Packed weights (~10.5 MB total) stay resident in L2.
// ---------------------------------------------------------------------------
__global__ void pack_w(const float* __restrict__ W, _Float16* __restrict__ Wp,
                       int N, long total)
{
    long idx = (long)blockIdx.x * blockDim.x + threadIdx.x;
    if (idx >= total) return;
    int  j    = (int)(idx & 15);
    long r    = idx >> 4;
    int  lane = (int)(r & 31);
    r >>= 5;
    int nt16 = N >> 4;
    int tn   = (int)(r % nt16);
    int kc   = (int)(r / nt16);
    int krow = kc * 32 + (j >> 3) * 16 + (lane >> 4) * 8 + (j & 7);
    int col  = tn * 16 + (lane & 15);
    Wp[idx] = (_Float16)W[(long)krow * N + col];
}

// ---------------------------------------------------------------------------
// WMMA GEMM: C[M,N] = act(A[M,K](lda) @ W[K,N] + bias)
// A: fp32 in memory, converted to f16 in-register (float4 vector loads).
// W: pre-packed f16 B-fragments (contiguous 32B load per fragment per lane).
// Wave tile = 32x64: 2 A-frags x 4 B-frags -> 8 v_wmma_f32_16x16x32_f16 per
// K-step, fp32 accumulation. One wave per block; EXEC all-ones around WMMA.
// act: 0=none 1=relu 2=softplus
// ---------------------------------------------------------------------------
__global__ __launch_bounds__(32)
void wmma_gemm(const float* __restrict__ A, const _Float16* __restrict__ Wp,
               const float* __restrict__ bias, float* __restrict__ C,
               int M, int N, int K, int lda, int act)
{
    const int lane = threadIdx.x;     // 0..31
    const int mr   = lane & 15;
    const int h    = lane >> 4;
    const int tileM = blockIdx.y;     // 32-row tile
    const int tileN = blockIdx.x;     // 64-col tile
    const int nt16  = N >> 4;

    const float* Arow0 = A + (size_t)(tileM * 32 + mr) * (size_t)lda;
    const float* Arow1 = Arow0 + (size_t)16 * lda;

    v8f acc[8] = {};   // acc[f] -> m-subtile 0, acc[4+f] -> m-subtile 1

    for (int k0 = 0; k0 < K; k0 += 32) {
        // ---- A fragments: two contiguous 32B runs per lane, cvt to f16 ----
        v16h a0, a1;
        {
            const float4* p0 = (const float4*)(Arow0 + k0 + h * 8);
            const float4* p1 = (const float4*)(Arow1 + k0 + h * 8);
            float af0[16], af1[16];
            *(float4*)(af0 + 0)  = p0[0];
            *(float4*)(af0 + 4)  = p0[1];
            *(float4*)(af0 + 8)  = p0[4];   // k0+16+h*8
            *(float4*)(af0 + 12) = p0[5];
            *(float4*)(af1 + 0)  = p1[0];
            *(float4*)(af1 + 4)  = p1[1];
            *(float4*)(af1 + 8)  = p1[4];
            *(float4*)(af1 + 12) = p1[5];
            #pragma unroll
            for (int j = 0; j < 16; ++j) {
                a0[j] = (_Float16)af0[j];
                a1[j] = (_Float16)af1[j];
            }
        }
        // ---- 4 packed B fragments, 8 WMMAs ----
        const _Float16* wbase =
            Wp + ((size_t)(k0 >> 5) * nt16 + (size_t)tileN * 4) * 512 + lane * 16;
        #pragma unroll
        for (int f = 0; f < 4; ++f) {
            v16h b = *(const v16h*)(wbase + f * 512);
            acc[f] = __builtin_amdgcn_wmma_f32_16x16x32_f16(
                false, a0, false, b, (short)0, acc[f], false, false);
            acc[4 + f] = __builtin_amdgcn_wmma_f32_16x16x32_f16(
                false, a1, false, b, (short)0, acc[4 + f], false, false);
        }
    }

    // ---- epilogue: bias + activation + store ----
    #pragma unroll
    for (int f = 0; f < 4; ++f) {
        const int col = tileN * 64 + f * 16 + mr;
        const float bv = bias ? bias[col] : 0.0f;
        #pragma unroll
        for (int mt = 0; mt < 2; ++mt) {
            v8f cacc = acc[mt * 4 + f];
            #pragma unroll
            for (int r = 0; r < 8; ++r) {
                float v = cacc[r] + bv;
                if (act == 1)      v = fmaxf(v, 0.0f);
                else if (act == 2) v = (v > 20.0f) ? v : log1pf(__expf(v));
                C[(size_t)(tileM * 32 + mt * 16 + r + 8 * h) * N + col] = v;
            }
        }
    }
}

// ---------------------------------------------------------------------------
// Causal depthwise conv (width 4) over time + SiLU.
// xz: (B*L, 2E) -- xs is cols [0,E). out: (B*L, E)
// ---------------------------------------------------------------------------
__global__ void conv_silu(const float* __restrict__ xz, const float* __restrict__ w,
                          const float* __restrict__ cb, float* __restrict__ out,
                          int L, int E, long total)
{
    long idx = (long)blockIdx.x * blockDim.x + threadIdx.x;
    if (idx >= total) return;
    int  e = (int)(idx % E);
    long r = idx / E;
    int  t = (int)(r % L);
    long b = r / L;

    float acc = cb[e];
    #pragma unroll
    for (int i = 0; i < D_CONV; ++i) {
        int ti = t - (D_CONV - 1) + i;
        if (ti >= 0)
            acc += w[i * E + e] * xz[(b * L + ti) * (size_t)(2 * E) + e];
    }
    out[idx] = acc / (1.0f + __expf(-acc));   // silu
}

// ---------------------------------------------------------------------------
// Selective scan. One block per (256 channels, batch). h[16] kept in regs.
// B_t/C_t (shared across channels) staged via LDS. Fuses +x*D skip and
// silu(z) gate; writes y in place over xs_y.
// ---------------------------------------------------------------------------
__global__ __launch_bounds__(256)
void scan_kernel(float* __restrict__ xs_y, const float* __restrict__ dt,
                 const float* __restrict__ xdbl, const float* __restrict__ xz,
                 const float* __restrict__ A_log, const float* __restrict__ Dp,
                 int L, int E)
{
    __shared__ float Bs[D_STATE];
    __shared__ float Cs[D_STATE];

    const int b = blockIdx.y;
    const int e = blockIdx.x * 256 + threadIdx.x;

    float Av[D_STATE];
    #pragma unroll
    for (int n = 0; n < D_STATE; ++n)
        Av[n] = -__expf(A_log[e * D_STATE + n]);

    float h[D_STATE];
    #pragma unroll
    for (int n = 0; n < D_STATE; ++n) h[n] = 0.0f;

    const float Dv = Dp[e];

    for (int t = 0; t < L; ++t) {
        const long r = (long)b * L + t;
        if (threadIdx.x < 32) {
            int n = threadIdx.x & 15;
            if (threadIdx.x < 16) Bs[n] = xdbl[r * 64 + DT_RANK + n];
            else                  Cs[n] = xdbl[r * 64 + DT_RANK + D_STATE + n];
        }
        __syncthreads();

        const float xv  = xs_y[r * E + e];
        const float dtv = dt[r * E + e];
        const float zv  = xz[r * (size_t)(2 * E) + E + e];
        const float dtx = dtv * xv;

        float y = 0.0f;
        #pragma unroll
        for (int n = 0; n < D_STATE; ++n) {
            float dA = __expf(dtv * Av[n]);
            h[n] = dA * h[n] + dtx * Bs[n];
            y += h[n] * Cs[n];
        }
        y = (y + xv * Dv) * (zv / (1.0f + __expf(-zv)));
        xs_y[r * E + e] = y;
        __syncthreads();
    }
}

// ---------------------------------------------------------------------------
// out = LayerNorm(x + y). One block (128 thr) per row, D = 512.
// y rows indexed in per-batch sequences of length Ly, taking the LAST LT rows.
// ---------------------------------------------------------------------------
__global__ __launch_bounds__(128)
void add_ln(const float* __restrict__ x, const float* __restrict__ y,
            const float* __restrict__ g, const float* __restrict__ be,
            float* __restrict__ out, int LT, int Ly)
{
    const int r = blockIdx.x;
    const int bidx = r / LT;
    const int t = r % LT;
    const long ry = (long)bidx * Ly + (Ly - LT) + t;

    __shared__ float red[128];
    __shared__ float mu_s, rstd_s;

    const int tid = threadIdx.x;
    float vals[4];
    float s = 0.0f;
    #pragma unroll
    for (int i = 0; i < 4; ++i) {
        int d = tid + i * 128;
        vals[i] = x[(long)r * D_MODEL + d] + y[ry * D_MODEL + d];
        s += vals[i];
    }
    red[tid] = s;
    __syncthreads();
    for (int off = 64; off > 0; off >>= 1) {
        if (tid < off) red[tid] += red[tid + off];
        __syncthreads();
    }
    if (tid == 0) mu_s = red[0] / (float)D_MODEL;
    __syncthreads();
    const float mu = mu_s;

    float v = 0.0f;
    #pragma unroll
    for (int i = 0; i < 4; ++i) { float d0 = vals[i] - mu; v += d0 * d0; }
    red[tid] = v;
    __syncthreads();
    for (int off = 64; off > 0; off >>= 1) {
        if (tid < off) red[tid] += red[tid + off];
        __syncthreads();
    }
    if (tid == 0) rstd_s = rsqrtf(red[0] / (float)D_MODEL + EPS_LN);
    __syncthreads();
    const float rstd = rstd_s;

    #pragma unroll
    for (int i = 0; i < 4; ++i) {
        int d = tid + i * 128;
        out[(long)r * D_MODEL + d] = (vals[i] - mu) * rstd * g[d] + be[d];
    }
}

// cat[b, 0:LS] = memory[b];  cat[b, LS:LS+LT] = t1[b]
__global__ void concat_kernel(const float* __restrict__ mem, const float* __restrict__ t1,
                              float* __restrict__ cat, long total)
{
    long idx = (long)blockIdx.x * blockDim.x + threadIdx.x;
    if (idx >= total) return;
    int  d = (int)(idx % D_MODEL);
    long r = idx / D_MODEL;
    int  t = (int)(r % L_CAT);
    long b = r / L_CAT;
    float v = (t < L_SRC)
        ? mem[(b * L_SRC + t) * (size_t)D_MODEL + d]
        : t1[(b * L_TGT + (t - L_SRC)) * (size_t)D_MODEL + d];
    cat[idx] = v;
}

static void launch_pack(const float* W, _Float16* Wp, int K, int N, hipStream_t s)
{
    long tot = (long)K * N;
    pack_w<<<(unsigned)((tot + 255) / 256), 256, 0, s>>>(W, Wp, N, tot);
}

// ---------------------------------------------------------------------------
// One full Mamba block: in_proj -> conv+silu -> x_proj -> dt -> scan -> out_proj
// Weight pointers are pre-packed f16 fragments.
// ---------------------------------------------------------------------------
static void run_mamba(const float* Xin, int Bn, int L,
                      const _Float16* p_in, const float* conv_w, const float* conv_b,
                      const _Float16* p_xp, const _Float16* p_dw, const float* dt_b,
                      const float* A_log, const float* Dp, const _Float16* p_op,
                      float* buf_xz, float* buf_conv, float* buf_xdbl, float* buf_dt,
                      float* outp, hipStream_t stream)
{
    const int E = D_INNER;
    const int M = Bn * L;
    dim3 blk(32);

    // xz = X @ in_proj   (M x 2E)
    wmma_gemm<<<dim3((2 * E) / 64, M / 32), blk, 0, stream>>>(
        Xin, p_in, nullptr, buf_xz, M, 2 * E, D_MODEL, D_MODEL, 0);

    // xs = silu(conv(xs) + b)
    long totc = (long)M * E;
    conv_silu<<<(unsigned)((totc + 255) / 256), 256, 0, stream>>>(
        buf_xz, conv_w, conv_b, buf_conv, L, E, totc);

    // x_dbl = xs @ x_proj  (M x 64)
    wmma_gemm<<<dim3(64 / 64, M / 32), blk, 0, stream>>>(
        buf_conv, p_xp, nullptr, buf_xdbl, M, 64, E, E, 0);

    // dt = softplus(x_dbl[:, :32] @ dt_w + dt_b)   (M x E)
    wmma_gemm<<<dim3(E / 64, M / 32), blk, 0, stream>>>(
        buf_xdbl, p_dw, dt_b, buf_dt, M, E, DT_RANK, 64, 2);

    // selective scan (y written in place over buf_conv, gated by silu(z))
    scan_kernel<<<dim3(E / 256, Bn), 256, 0, stream>>>(
        buf_conv, buf_dt, buf_xdbl, buf_xz, A_log, Dp, L, E);

    // out = y @ out_proj  (M x D_MODEL)
    wmma_gemm<<<dim3(D_MODEL / 64, M / 32), blk, 0, stream>>>(
        buf_conv, p_op, nullptr, outp, M, D_MODEL, E, E, 0);
}

extern "C" void kernel_launch(void* const* d_in, const int* in_sizes, int n_in,
                              void* d_out, int out_size, void* d_ws, size_t ws_size,
                              hipStream_t stream)
{
    (void)in_sizes; (void)n_in; (void)out_size; (void)ws_size;
    auto F = [&](int i) { return (const float*)d_in[i]; };

    const float* tgt    = F(0);
    const float* memory = F(1);
    // self mamba params: 2..10 ; cross: 11..19 (dict order of _mamba_params)
    const float *s_in = F(2),  *s_cw = F(3),  *s_cb = F(4),  *s_xp = F(5),
                *s_dw = F(6),  *s_db = F(7),  *s_al = F(8),  *s_D = F(9),  *s_op = F(10);
    const float *c_in = F(11), *c_cw = F(12), *c_cb = F(13), *c_xp = F(14),
                *c_dw = F(15), *c_db = F(16), *c_al = F(17), *c_D = F(18), *c_op = F(19);
    const float *ffn_w1 = F(20), *ffn_b1 = F(21), *ffn_w2 = F(22), *ffn_b2 = F(23);
    const float *g1 = F(24), *be1 = F(25), *g2 = F(26), *be2 = F(27), *g3 = F(28), *be3 = F(29);

    const long BT_T = (long)BATCH * L_TGT;   // 8192
    const long BT_C = (long)BATCH * L_CAT;   // 24576

    // Workspace layout (fp32 activations sized for the cross phase)
    float* ws       = (float*)d_ws;
    float* buf_xz   = ws;                              // BT_C * 2048
    float* buf_conv = buf_xz   + BT_C * 2048L;         // BT_C * 1024
    float* buf_xdbl = buf_conv + BT_C * 1024L;         // BT_C * 64
    float* buf_dt   = buf_xdbl + BT_C * 64L;           // BT_C * 1024
    float* buf_m    = buf_dt   + BT_C * 1024L;         // BT_C * 512
    float* buf_t1   = buf_m    + BT_C * 512L;          // BT_T * 512
    float* buf_t2   = buf_t1   + BT_T * 512L;          // BT_T * 512
    float* buf_cat  = buf_t2   + BT_T * 512L;          // BT_C * 512

    // Packed f16 weights (B-fragment layout), after the fp32 region
    _Float16* hp   = (_Float16*)(buf_cat + BT_C * 512L);
    _Float16* ps_in = hp;                               // 512*2048
    _Float16* ps_xp = ps_in + 512L * 2048;              // 1024*64
    _Float16* ps_dw = ps_xp + 1024L * 64;               // 32*1024
    _Float16* ps_op = ps_dw + 32L * 1024;               // 1024*512
    _Float16* pc_in = ps_op + 1024L * 512;
    _Float16* pc_xp = pc_in + 512L * 2048;
    _Float16* pc_dw = pc_xp + 1024L * 64;
    _Float16* pc_op = pc_dw + 32L * 1024;
    _Float16* pf_w1 = pc_op + 1024L * 512;              // 512*2048
    _Float16* pf_w2 = pf_w1 + 512L * 2048;              // 2048*512

    // ---- pack all weights to f16 WMMA fragments (once per launch) ----
    launch_pack(s_in, ps_in, D_MODEL, 2 * D_INNER, stream);
    launch_pack(s_xp, ps_xp, D_INNER, DT_RANK + 2 * D_STATE, stream);
    launch_pack(s_dw, ps_dw, DT_RANK, D_INNER, stream);
    launch_pack(s_op, ps_op, D_INNER, D_MODEL, stream);
    launch_pack(c_in, pc_in, D_MODEL, 2 * D_INNER, stream);
    launch_pack(c_xp, pc_xp, D_INNER, DT_RANK + 2 * D_STATE, stream);
    launch_pack(c_dw, pc_dw, DT_RANK, D_INNER, stream);
    launch_pack(c_op, pc_op, D_INNER, D_MODEL, stream);
    launch_pack(ffn_w1, pf_w1, D_MODEL, D_FFN, stream);
    launch_pack(ffn_w2, pf_w2, D_FFN, D_MODEL, stream);

    float* outp = (float*)d_out;

    // ---- self mamba + residual + LN1 ----
    run_mamba(tgt, BATCH, L_TGT,
              ps_in, s_cw, s_cb, ps_xp, ps_dw, s_db, s_al, s_D, ps_op,
              buf_xz, buf_conv, buf_xdbl, buf_dt, buf_m, stream);
    add_ln<<<(unsigned)BT_T, 128, 0, stream>>>(tgt, buf_m, g1, be1, buf_t1, L_TGT, L_TGT);

    // ---- cross mamba over [memory; tgt], tail slice, + residual + LN2 ----
    long totcat = BT_C * (long)D_MODEL;
    concat_kernel<<<(unsigned)((totcat + 255) / 256), 256, 0, stream>>>(
        memory, buf_t1, buf_cat, totcat);
    run_mamba(buf_cat, BATCH, L_CAT,
              pc_in, c_cw, c_cb, pc_xp, pc_dw, c_db, c_al, c_D, pc_op,
              buf_xz, buf_conv, buf_xdbl, buf_dt, buf_m, stream);
    add_ln<<<(unsigned)BT_T, 128, 0, stream>>>(buf_t1, buf_m, g2, be2, buf_t2, L_TGT, L_CAT);

    // ---- FFN + residual + LN3 ----
    dim3 blk(32);
    wmma_gemm<<<dim3(D_FFN / 64, (int)(BT_T / 32)), blk, 0, stream>>>(
        buf_t2, pf_w1, ffn_b1, buf_xz, (int)BT_T, D_FFN, D_MODEL, D_MODEL, 1);
    wmma_gemm<<<dim3(D_MODEL / 64, (int)(BT_T / 32)), blk, 0, stream>>>(
        buf_xz, pf_w2, ffn_b2, buf_m, (int)BT_T, D_MODEL, D_FFN, D_FFN, 0);
    add_ln<<<(unsigned)BT_T, 128, 0, stream>>>(buf_t2, buf_m, g3, be3, outp, L_TGT, L_TGT);
}
